// SparseSelfAttention_75960791597210
// MI455X (gfx1250) — matrix operand order repaired and checked by reading, not compiled, not run on hardware
//
#include <hip/hip_runtime.h>
#include <hip/hip_bf16.h>

// ---------------------------------------------------------------------------
// Windowed self-attention for MI455X (gfx1250, wave32, WMMA).
//   B=2, S=2048, D_MODEL=1024, H=16, DEPTH=64, WINDOW=128.
//
// v3: GEMM waves now compute 2 M-tiles x 4 N-tiles (24 WMMAs per 32-wide K
// step, each B fragment reused across two A fragments). All f32->f16 hi/lo
// splitting remains hoisted into one-time pre-passes with WMMA-fragment-
// friendly layouts:
//   - x       -> xh/xl        row-major   [M][K]      (A fragments)
//   - W*      -> w*t_h/_l     transposed  [N][K]      (B fragments)
//   - Q,K     -> hi/lo f16    head-split  [B,H,S,64]  (attention QK^T)
//   - V       -> hi/lo f16    head-transp [B,H,64,S]  (attention P@V)
//   - ctx     -> hi/lo f16    row-major   (reuses the x region)
// Workspace: 40M halfs = 80 MB, fully L2-resident (192 MB L2).
// ---------------------------------------------------------------------------

typedef __attribute__((ext_vector_type(16))) _Float16 v16h;
typedef __attribute__((ext_vector_type(8)))  _Float16 h8;
typedef __attribute__((ext_vector_type(8)))  float    v8f;

#define D_MODEL   1024
#define NUM_HEADS 16
#define DEPTH     64
#define WINDOW    128
#define SEQ       2048
#define BATCH     2
#define MROWS     (BATCH * SEQ)       // 4096
#define MAXC      288                 // 17 key tiles (272) padded to 9*32

union Frag { v16h v; h8 h[2]; };

__device__ __forceinline__ v8f wmma_f16(v16h a, v16h b, v8f c) {
  return __builtin_amdgcn_wmma_f32_16x16x32_f16(false, a, false, b,
                                                (short)0, c, false, false);
}

// ---------------------------------------------------------------------------
// Pre-pass: elementwise split f32 -> f16 hi + f16 residual.
// ---------------------------------------------------------------------------
__global__ __launch_bounds__(256) void split_f32(
    const float* __restrict__ src, _Float16* __restrict__ h,
    _Float16* __restrict__ l, int n)
{
  const int i = blockIdx.x * 256 + threadIdx.x;
  if (i < n) {
    const float x = src[i];
    const _Float16 t = (_Float16)x;
    h[i] = t;
    l[i] = (_Float16)(x - (float)t);
  }
}

// Pre-pass: split + transpose W[K][N] -> th/tl[N][K].
__global__ __launch_bounds__(256) void split_w_t(
    const float* __restrict__ W, _Float16* __restrict__ th,
    _Float16* __restrict__ tl, int K, int N)
{
  const int i = blockIdx.x * 256 + threadIdx.x;
  if (i < K * N) {
    const int k = i / N, n = i - k * N;
    const float x = W[i];
    const _Float16 t = (_Float16)x;
    const size_t o = (size_t)n * K + k;
    th[o] = t;
    tl[o] = (_Float16)(x - (float)t);
  }
}

// ---------------------------------------------------------------------------
// GEMM: C = A[M,K] @ B[K,N] + bias, A row-major hi/lo f16, B transposed
// [N][K] hi/lo f16. One wave computes 2 M-tiles x 4 N-tiles (32x64 of C):
// per 32-wide K step: 8 A-frag loads + 16 B-frag loads -> 24 WMMAs.
// mode 0: f32 row-major [M,N] -> dst0
// mode 1: f16 hi/lo head-split  [B,H,S,64]  -> dst0/dst1
// mode 2: f16 hi/lo head-transp [B,H,64,S]  -> dst0/dst1
// ---------------------------------------------------------------------------
__global__ __launch_bounds__(128) void gemm16p(
    const _Float16* __restrict__ Ah, const _Float16* __restrict__ Al,
    const _Float16* __restrict__ Bh, const _Float16* __restrict__ Bl,
    const float* __restrict__ bias,
    void* __restrict__ dst0, void* __restrict__ dst1,
    int M, int N, int Kd, int mode)
{
  const int lane = threadIdx.x & 31;
  const int warp = threadIdx.x >> 5;
  const int wid  = blockIdx.x * 4 + warp;
  const int ngrp = N >> 6;                 // groups of 4 N-tiles
  const int tm2  = wid / ngrp;             // 32-row super-tile
  const int tn0  = (wid - tm2 * ngrp) * 4;
  if (tm2 * 32 >= M) return;

  const int m  = lane & 15;
  const int hi = lane >> 4;

  const _Float16* __restrict__ arow_h0 = Ah + (size_t)(tm2 * 32 + m) * Kd;
  const _Float16* __restrict__ arow_l0 = Al + (size_t)(tm2 * 32 + m) * Kd;
  const _Float16* __restrict__ arow_h1 = arow_h0 + (size_t)16 * Kd;
  const _Float16* __restrict__ arow_l1 = arow_l0 + (size_t)16 * Kd;

  v8f acc0[4] = {{}, {}, {}, {}};
  v8f acc1[4] = {{}, {}, {}, {}};
  for (int k0 = 0; k0 < Kd; k0 += 32) {
    const int ka = k0 + hi * 8;            // A K-pattern: [ka,ka+8) U [ka+16,ka+24)
    Frag a0h, a0l, a1h, a1l;
    a0h.h[0] = *(const h8*)(arow_h0 + ka);
    a0h.h[1] = *(const h8*)(arow_h0 + ka + 16);
    a0l.h[0] = *(const h8*)(arow_l0 + ka);
    a0l.h[1] = *(const h8*)(arow_l0 + ka + 16);
    a1h.h[0] = *(const h8*)(arow_h1 + ka);
    a1h.h[1] = *(const h8*)(arow_h1 + ka + 16);
    a1l.h[0] = *(const h8*)(arow_l1 + ka);
    a1l.h[1] = *(const h8*)(arow_l1 + ka + 16);

    const int kb = k0 + hi * 8;            // B K-pattern: [kb,kb+8) U [kb+16,kb+24)
#pragma unroll
    for (int nt = 0; nt < 4; ++nt) {
      const size_t co = (size_t)((tn0 + nt) * 16 + m) * Kd + kb;
      Frag bf_h, bf_l;
      bf_h.h[0] = *(const h8*)(Bh + co);
      bf_h.h[1] = *(const h8*)(Bh + co + 16);
      bf_l.h[0] = *(const h8*)(Bl + co);
      bf_l.h[1] = *(const h8*)(Bl + co + 16);
      acc0[nt] = wmma_f16(a0h.v, bf_h.v, acc0[nt]);
      acc0[nt] = wmma_f16(a0h.v, bf_l.v, acc0[nt]);
      acc0[nt] = wmma_f16(a0l.v, bf_h.v, acc0[nt]);
      acc1[nt] = wmma_f16(a1h.v, bf_h.v, acc1[nt]);
      acc1[nt] = wmma_f16(a1h.v, bf_l.v, acc1[nt]);
      acc1[nt] = wmma_f16(a1l.v, bf_h.v, acc1[nt]);
    }
  }

#pragma unroll
  for (int g = 0; g < 2; ++g) {
    const v8f* acc = g ? acc1 : acc0;
#pragma unroll
    for (int nt = 0; nt < 4; ++nt) {
      const float bv = bias[(tn0 + nt) * 16 + m];
#pragma unroll
      for (int r = 0; r < 8; ++r) {
        const int row = tm2 * 32 + g * 16 + r + 8 * hi;
        const int col = (tn0 + nt) * 16 + m;
        const float val = acc[nt][r] + bv;
        if (mode == 0) {
          ((float*)dst0)[(size_t)row * N + col] = val;
        } else {
          const int b = row >> 11, s = row & (SEQ - 1);
          const int h = col >> 6,  d = col & (DEPTH - 1);
          const size_t idx = (mode == 1)
              ? (((size_t)(b * NUM_HEADS + h)) * SEQ + s) * DEPTH + d
              : (((size_t)(b * NUM_HEADS + h)) * DEPTH + d) * SEQ + s;
          const _Float16 t = (_Float16)val;
          ((_Float16*)dst0)[idx] = t;
          ((_Float16*)dst1)[idx] = (_Float16)(val - (float)t);
        }
      }
    }
  }
}

// ---------------------------------------------------------------------------
// Windowed attention: one wave per (b, h, 16-query tile). All operands f16.
// ---------------------------------------------------------------------------
__global__ __launch_bounds__(32) void attn_win(
    const _Float16* __restrict__ Qh, const _Float16* __restrict__ Ql,
    const _Float16* __restrict__ Kh, const _Float16* __restrict__ Kl,
    const _Float16* __restrict__ Vth, const _Float16* __restrict__ Vtl,
    _Float16* __restrict__ ctxh, _Float16* __restrict__ ctxl)
{
  __shared__ alignas(16) float    sc[16][MAXC];
  __shared__ alignas(16) _Float16 pr[16][MAXC];
  __shared__ float rowsum[16];

  const int lane = threadIdx.x;
  const int tile = blockIdx.x;
  const int qt   = tile & 127;
  const int bh   = tile >> 7;
  const int q0   = qt * 16;

  const int kt_lo = (qt - 8 > 0) ? qt - 8 : 0;
  const int kt_hi = (qt + 8 < 127) ? qt + 8 : 127;
  const int nkt   = kt_hi - kt_lo + 1;   // <= 17
  const int ncols = nkt * 16;            // <= 272

  const int m  = lane & 15;
  const int hi = lane >> 4;

  const size_t hb = (size_t)bh * SEQ * DEPTH;

  // ---------------- Phase 1: score tiles (3 WMMAs / 32-K step) ----------
  for (int j = 0; j < nkt; ++j) {
    const int key0 = (kt_lo + j) * 16;
    v8f acc = {};
#pragma unroll
    for (int ks = 0; ks < DEPTH; ks += 32) {
      Frag qa_h, qa_l;
      const size_t qo = hb + (size_t)(q0 + m) * DEPTH + ks + hi * 8;
      qa_h.h[0] = *(const h8*)(Qh + qo);
      qa_h.h[1] = *(const h8*)(Qh + qo + 16);
      qa_l.h[0] = *(const h8*)(Ql + qo);
      qa_l.h[1] = *(const h8*)(Ql + qo + 16);

      Frag kb_h, kb_l;                   // B = K^T, same depth pattern as Q
      const size_t ko = hb + (size_t)(key0 + m) * DEPTH + ks + hi * 8;
      kb_h.h[0] = *(const h8*)(Kh + ko);
      kb_h.h[1] = *(const h8*)(Kh + ko + 16);
      kb_l.h[0] = *(const h8*)(Kl + ko);
      kb_l.h[1] = *(const h8*)(Kl + ko + 16);

      acc = wmma_f16(qa_h.v, kb_h.v, acc);
      acc = wmma_f16(qa_h.v, kb_l.v, acc);
      acc = wmma_f16(qa_l.v, kb_h.v, acc);
    }
#pragma unroll
    for (int r = 0; r < 8; ++r) {
      const int mm = r + 8 * hi;
      const int qi = q0 + mm, ki = key0 + m;
      float s = acc[r] * 0.125f;         // 1/sqrt(64)
      int d = ki - qi; if (d < 0) d = -d;
      if (d > WINDOW) s = -1.0e9f;
      sc[mm][j * 16 + m] = s;
    }
  }
  __syncthreads();

  for (int idx = lane; idx < 16 * MAXC; idx += 32)
    (&pr[0][0])[idx] = (_Float16)0.0f;
  __syncthreads();

  // ---------------- Softmax ----------------
  for (int row = 0; row < 16; ++row) {
    float mx = -3.0e38f;
    for (int c = lane; c < ncols; c += 32) mx = fmaxf(mx, sc[row][c]);
#pragma unroll
    for (int off = 16; off; off >>= 1) mx = fmaxf(mx, __shfl_xor(mx, off, 32));
    float sum = 0.0f;
    for (int c = lane; c < ncols; c += 32) {
      const float p = __expf(sc[row][c] - mx);
      pr[row][c] = (_Float16)p;
      sum += p;
    }
#pragma unroll
    for (int off = 16; off; off >>= 1) sum += __shfl_xor(sum, off, 32);
    if (lane == 0) rowsum[row] = sum;
  }
  __syncthreads();

  // ---------------- Phase 2: out = P @ V (2 WMMAs / 32-K step) ----------
  const int nks = (ncols + 31) >> 5;     // <= 9
  const int bb  = bh >> 4, hh = bh & 15;
#pragma unroll
  for (int t = 0; t < 4; ++t) {
    v8f acc = {};
    for (int ks = 0; ks < nks; ++ks) {
      const int kb = ks * 32;
      Frag pa;                            // probs A fragment from LDS
      pa.h[0] = *(const h8*)(&pr[m][kb + hi * 16]);
      pa.h[1] = *(const h8*)(&pr[m][kb + hi * 16 + 8]);

      Frag vf_h, vf_l;                    // V^T fragment: contiguous keys
      const size_t vrow = ((size_t)bh * DEPTH + t * 16 + m) * SEQ;
      const int kstart = kt_lo * 16 + kb + hi * 16;
      if (kstart + 15 < SEQ) {
        vf_h.h[0] = *(const h8*)(Vth + vrow + kstart);
        vf_h.h[1] = *(const h8*)(Vth + vrow + kstart + 8);
        vf_l.h[0] = *(const h8*)(Vtl + vrow + kstart);
        vf_l.h[1] = *(const h8*)(Vtl + vrow + kstart + 8);
      } else {
        // padded tail: probs are zero there, clamp reads into range
#pragma unroll
        for (int e = 0; e < 16; ++e) {
          int key = kstart + e; if (key > SEQ - 1) key = SEQ - 1;
          vf_h.v[e] = Vth[vrow + key];
          vf_l.v[e] = Vtl[vrow + key];
        }
      }
      acc = wmma_f16(pa.v, vf_h.v, acc);
      acc = wmma_f16(pa.v, vf_l.v, acc);
    }
#pragma unroll
    for (int r = 0; r < 8; ++r) {
      const int mm = r + 8 * hi;
      const float o = acc[r] / rowsum[mm];
      const size_t idx =
          ((size_t)(bb * SEQ + q0 + mm)) * D_MODEL + hh * DEPTH + t * 16 + m;
      const _Float16 th = (_Float16)o;
      ctxh[idx] = th;
      ctxl[idx] = (_Float16)(o - (float)th);
    }
  }
}

// ---------------------------------------------------------------------------
extern "C" void kernel_launch(void* const* d_in, const int* in_sizes, int n_in,
                              void* d_out, int out_size, void* d_ws, size_t ws_size,
                              hipStream_t stream) {
  (void)in_sizes; (void)n_in; (void)out_size; (void)ws_size;
  const float* x  = (const float*)d_in[0];
  const float* Wq = (const float*)d_in[1];
  const float* bq = (const float*)d_in[2];
  const float* Wk = (const float*)d_in[3];
  const float* bk = (const float*)d_in[4];
  const float* Wv = (const float*)d_in[5];
  const float* bv = (const float*)d_in[6];
  const float* Wo = (const float*)d_in[7];
  const float* bo = (const float*)d_in[8];

  const size_t T = (size_t)MROWS * D_MODEL;       // 4M elems (x / Q / K / V / ctx)
  const size_t WSZ = (size_t)D_MODEL * D_MODEL;   // 1M elems (weights)
  _Float16* base = (_Float16*)d_ws;
  _Float16* xh   = base;            _Float16* xl   = xh + T;     // reused as ctx
  _Float16* wqh  = xl  + T;         _Float16* wql  = wqh + WSZ;
  _Float16* wkh  = wql + WSZ;       _Float16* wkl  = wkh + WSZ;
  _Float16* wvh  = wkl + WSZ;       _Float16* wvl  = wvh + WSZ;
  _Float16* woh  = wvl + WSZ;       _Float16* wol  = woh + WSZ;
  _Float16* qh   = wol + WSZ;       _Float16* ql   = qh  + T;
  _Float16* kh   = ql  + T;         _Float16* kl   = kh  + T;
  _Float16* vth  = kl  + T;         _Float16* vtl  = vth + T;
  _Float16* ctxh = xh;              _Float16* ctxl = xl;          // alias x

  // --- pre-passes: split / transpose (memory-bound, one-time) ---
  split_f32<<<dim3((int)(T / 256)), dim3(256), 0, stream>>>(x, xh, xl, (int)T);
  split_w_t<<<dim3((int)(WSZ / 256)), dim3(256), 0, stream>>>(Wq, wqh, wql, D_MODEL, D_MODEL);
  split_w_t<<<dim3((int)(WSZ / 256)), dim3(256), 0, stream>>>(Wk, wkh, wkl, D_MODEL, D_MODEL);
  split_w_t<<<dim3((int)(WSZ / 256)), dim3(256), 0, stream>>>(Wv, wvh, wvl, D_MODEL, D_MODEL);
  split_w_t<<<dim3((int)(WSZ / 256)), dim3(256), 0, stream>>>(Wo, woh, wol, D_MODEL, D_MODEL);

  // --- QKV projections: (4096/32) M-supertiles x 16 N-groups = 2048 waves ---
  dim3 ggrid(512), gblk(128);
  gemm16p<<<ggrid, gblk, 0, stream>>>(xh, xl, wqh, wql, bq, qh,  ql,  MROWS, D_MODEL, D_MODEL, 1);
  gemm16p<<<ggrid, gblk, 0, stream>>>(xh, xl, wkh, wkl, bk, kh,  kl,  MROWS, D_MODEL, D_MODEL, 1);
  gemm16p<<<ggrid, gblk, 0, stream>>>(xh, xl, wvh, wvl, bv, vth, vtl, MROWS, D_MODEL, D_MODEL, 2);

  // --- windowed attention: 2*16*128 = 4096 tiles ---
  attn_win<<<dim3(4096), dim3(32), 0, stream>>>(qh, ql, kh, kl, vth, vtl, ctxh, ctxl);

  // --- output projection -> d_out (f32) ---
  gemm16p<<<ggrid, gblk, 0, stream>>>(ctxh, ctxl, woh, wol, bo, d_out, nullptr,
                                      MROWS, D_MODEL, D_MODEL, 0);
}
